// RvNN_25383256719784
// MI455X (gfx1250) — matrix operand order repaired and checked by reading, not compiled, not run on hardware
//
#include <hip/hip_runtime.h>
#include <hip/hip_bf16.h>

// ---------------------------------------------------------------------------
// Problem constants
// ---------------------------------------------------------------------------
#define C_CH   8
#define DEPTH  5
#define E_DIM  64
#define H_DIM  512
#define G_DIM  1536   // 3*H

static const int SIZES[6]   = {1, 8, 64, 512, 4096, 32768};
static const int OFFSETS[7] = {0, 1, 9, 73, 585, 4681, 37449};

// ---------------------------------------------------------------------------
// Types / helpers
// ---------------------------------------------------------------------------
typedef __attribute__((ext_vector_type(16))) __bf16          v16bf;
typedef __attribute__((ext_vector_type(8)))  float           v8f;
typedef __attribute__((ext_vector_type(8)))  unsigned short  u16x8;
typedef __attribute__((ext_vector_type(16))) unsigned short  u16x16;

__device__ __forceinline__ unsigned short f2bfu(float f) {
    unsigned u = __builtin_bit_cast(unsigned, f);
    unsigned r = u + 0x7FFFu + ((u >> 16) & 1u);   // round-to-nearest-even
    return (unsigned short)(r >> 16);
}
__device__ __forceinline__ float bfu2f(unsigned short h) {
    return __builtin_bit_cast(float, (unsigned)h << 16);
}
__device__ __forceinline__ float sigmoidf_(float x) {
    return 1.0f / (1.0f + __expf(-x));
}

// A fragment: 16x32 bf16, lane<16 -> M=lane, K = {k0..k0+7, k0+16..k0+23}
//             lane>=16 -> M=lane-16, K = {k0+8..k0+15, k0+24..k0+31}
__device__ __forceinline__ v16bf load_frag_a(const unsigned short* __restrict__ row,
                                             int k0, int ksel) {
    u16x8 lo = *(const u16x8*)(row + k0 + ksel);
    u16x8 hi = *(const u16x8*)(row + k0 + 16 + ksel);
    u16x16 c;
#pragma unroll
    for (int i = 0; i < 8; ++i) { c[i] = lo[i]; c[i + 8] = hi[i]; }
    return __builtin_bit_cast(v16bf, c);
}

// B fragment: 32x16 bf16 (we feed W row-major NxK, lane = column N = lane&15),
// lane<16 -> K = {k0..k0+7, k0+16..k0+23}, lane>=16 -> K = {k0+8..k0+15, k0+24..k0+31}.
__device__ __forceinline__ v16bf load_frag_b(const unsigned short* __restrict__ row,
                                             int kb) {
    u16x8 lo = *(const u16x8*)(row + kb);
    u16x8 hi = *(const u16x8*)(row + kb + 16);
    u16x16 c;
#pragma unroll
    for (int i = 0; i < 8; ++i) { c[i] = lo[i]; c[i + 8] = hi[i]; }
    return __builtin_bit_cast(v16bf, c);
}

// ---------------------------------------------------------------------------
// Generic WMMA GEMM:  out = act( A1 * W1^T [+ A2 * W2^T] + bias )
//   A1: bf16, row r data at  gather? gbase + gather[r]*K1 : A1 + (r*mul+add)*K1
//   A2: bf16, row r data at  A2 + r*K2   (optional, pass nullptr)
//   W1/W2: bf16, N x K row-major
//   out: fp32 (outF) or bf16 (outB), M x N row-major
// One wave computes a 32(M) x 64(N) strip: 2 A frags x 4 B frags = 8 WMMAs per
// K-step of 32.  All 12 B128 loads are issued before the 8 WMMAs so the
// compiler emits one load clause + one wait per 8 matrix ops.
// ---------------------------------------------------------------------------
__global__ __launch_bounds__(256) void wmma_gemm_kernel(
    const unsigned short* __restrict__ A1, int a1_mul, int a1_add,
    const int* __restrict__ gather, const unsigned short* __restrict__ gbase,
    int K1, const unsigned short* __restrict__ W1,
    const unsigned short* __restrict__ A2, int K2,
    const unsigned short* __restrict__ W2,
    const float* __restrict__ bias,
    float* __restrict__ outF, unsigned short* __restrict__ outB,
    int M, int N, int act_tanh)
{
    const int gwave  = (int)((blockIdx.x * 256u + threadIdx.x) >> 5);
    const int lane   = (int)(threadIdx.x & 31u);
    const int tilesN = N >> 6;                 // N is a multiple of 64
    const int tilesM = (M + 31) >> 5;          // 32 rows per wave
    if (gwave >= tilesM * tilesN) return;
    const int tm = gwave / tilesN;
    const int tn = gwave - tm * tilesN;

    const int ksel  = (lane >> 4) << 3;        // 0 or 8   (A packing)
    const int kselB = (lane >> 4) << 4;        // 0 or 16  (unused; B packing also uses ksel)

    // two A-row pointers (row halves h=0,1), clamped; garbage rows never stored
    const unsigned short* arow[2];
#pragma unroll
    for (int h = 0; h < 2; ++h) {
        int rl = tm * 32 + h * 16 + (lane & 15);
        int rc = (rl < M) ? rl : (M - 1);
        if (gather) arow[h] = gbase + (size_t)gather[rc] * (size_t)K1;
        else        arow[h] = A1 + ((size_t)rc * (size_t)a1_mul + (size_t)a1_add) * (size_t)K1;
    }

    const unsigned short* wrow[4];
#pragma unroll
    for (int b = 0; b < 4; ++b)
        wrow[b] = W1 + (size_t)(tn * 64 + b * 16 + (lane & 15)) * (size_t)K1;

    v8f acc[2][4] = {};

    for (int k0 = 0; k0 < K1; k0 += 32) {
        v16bf a0 = load_frag_a(arow[0], k0, ksel);
        v16bf a1 = load_frag_a(arow[1], k0, ksel);
        v16bf w0 = load_frag_b(wrow[0], k0 + ksel);
        v16bf w1 = load_frag_b(wrow[1], k0 + ksel);
        v16bf w2 = load_frag_b(wrow[2], k0 + ksel);
        v16bf w3 = load_frag_b(wrow[3], k0 + ksel);
        acc[0][0] = __builtin_amdgcn_wmma_f32_16x16x32_bf16(false, a0, false, w0, (short)0, acc[0][0], false, false);
        acc[0][1] = __builtin_amdgcn_wmma_f32_16x16x32_bf16(false, a0, false, w1, (short)0, acc[0][1], false, false);
        acc[0][2] = __builtin_amdgcn_wmma_f32_16x16x32_bf16(false, a0, false, w2, (short)0, acc[0][2], false, false);
        acc[0][3] = __builtin_amdgcn_wmma_f32_16x16x32_bf16(false, a0, false, w3, (short)0, acc[0][3], false, false);
        acc[1][0] = __builtin_amdgcn_wmma_f32_16x16x32_bf16(false, a1, false, w0, (short)0, acc[1][0], false, false);
        acc[1][1] = __builtin_amdgcn_wmma_f32_16x16x32_bf16(false, a1, false, w1, (short)0, acc[1][1], false, false);
        acc[1][2] = __builtin_amdgcn_wmma_f32_16x16x32_bf16(false, a1, false, w2, (short)0, acc[1][2], false, false);
        acc[1][3] = __builtin_amdgcn_wmma_f32_16x16x32_bf16(false, a1, false, w3, (short)0, acc[1][3], false, false);
    }

    if (A2) {
        const unsigned short* arow2[2];
#pragma unroll
        for (int h = 0; h < 2; ++h) {
            int rl = tm * 32 + h * 16 + (lane & 15);
            int rc = (rl < M) ? rl : (M - 1);
            arow2[h] = A2 + (size_t)rc * (size_t)K2;
        }
        const unsigned short* wrow2[4];
#pragma unroll
        for (int b = 0; b < 4; ++b)
            wrow2[b] = W2 + (size_t)(tn * 64 + b * 16 + (lane & 15)) * (size_t)K2;
        for (int k0 = 0; k0 < K2; k0 += 32) {
            v16bf a0 = load_frag_a(arow2[0], k0, ksel);
            v16bf a1 = load_frag_a(arow2[1], k0, ksel);
            v16bf w0 = load_frag_b(wrow2[0], k0 + ksel);
            v16bf w1 = load_frag_b(wrow2[1], k0 + ksel);
            v16bf w2 = load_frag_b(wrow2[2], k0 + ksel);
            v16bf w3 = load_frag_b(wrow2[3], k0 + ksel);
            acc[0][0] = __builtin_amdgcn_wmma_f32_16x16x32_bf16(false, a0, false, w0, (short)0, acc[0][0], false, false);
            acc[0][1] = __builtin_amdgcn_wmma_f32_16x16x32_bf16(false, a0, false, w1, (short)0, acc[0][1], false, false);
            acc[0][2] = __builtin_amdgcn_wmma_f32_16x16x32_bf16(false, a0, false, w2, (short)0, acc[0][2], false, false);
            acc[0][3] = __builtin_amdgcn_wmma_f32_16x16x32_bf16(false, a0, false, w3, (short)0, acc[0][3], false, false);
            acc[1][0] = __builtin_amdgcn_wmma_f32_16x16x32_bf16(false, a1, false, w0, (short)0, acc[1][0], false, false);
            acc[1][1] = __builtin_amdgcn_wmma_f32_16x16x32_bf16(false, a1, false, w1, (short)0, acc[1][1], false, false);
            acc[1][2] = __builtin_amdgcn_wmma_f32_16x16x32_bf16(false, a1, false, w2, (short)0, acc[1][2], false, false);
            acc[1][3] = __builtin_amdgcn_wmma_f32_16x16x32_bf16(false, a1, false, w3, (short)0, acc[1][3], false, false);
        }
    }

    // C/D layout: lane L holds col N = (L&15); VGPR v holds row M = v + 8*(L>=16)
    const int cl = tn * 64 + (lane & 15);
#pragma unroll
    for (int h = 0; h < 2; ++h) {
        const int rbase = tm * 32 + h * 16 + ((lane >> 4) << 3);
#pragma unroll
        for (int b = 0; b < 4; ++b) {
            const int col = cl + b * 16;
            const float bv = bias ? bias[col] : 0.0f;
#pragma unroll
            for (int v = 0; v < 8; ++v) {
                const int rr = rbase + v;
                if (rr < M) {
                    float x = acc[h][b][v] + bv;
                    if (act_tanh) x = tanhf(x);
                    const size_t o = (size_t)rr * (size_t)N + (size_t)col;
                    if (outF) outF[o] = x;
                    else      outB[o] = f2bfu(x);
                }
            }
        }
    }
}

// ---------------------------------------------------------------------------
// GRU gate elementwise: h <- (1-z)*tanh(xn + r*hn) + z*h   (PyTorch convention)
// gx, gh: n x 1536 fp32 ; h: n x 512 bf16 (in-place, each thread owns its slot)
// ---------------------------------------------------------------------------
__global__ __launch_bounds__(256) void gru_gates_kernel(
    const float* __restrict__ gx, const float* __restrict__ gh,
    unsigned short* __restrict__ h, int total)
{
    const int idx = (int)(blockIdx.x * 256u + threadIdx.x);
    if (idx >= total) return;
    const int i = idx >> 9;          // / 512
    const int j = idx & 511;
    const float* gxr = gx + (size_t)i * G_DIM;
    const float* ghr = gh + (size_t)i * G_DIM;
    const float r    = sigmoidf_(gxr[j]        + ghr[j]);
    const float z    = sigmoidf_(gxr[512 + j]  + ghr[512 + j]);
    const float cand = tanhf(gxr[1024 + j] + r * ghr[1024 + j]);
    const float hp   = bfu2f(h[idx]);
    h[idx] = f2bfu((1.0f - z) * cand + z * hp);
}

// fp32 (strided) -> bf16 convert
__global__ __launch_bounds__(256) void convert_bf16_kernel(
    const float* __restrict__ src, int sld, int soff,
    unsigned short* __restrict__ dst, int rows, int cols)
{
    const int idx = (int)(blockIdx.x * 256u + threadIdx.x);
    if (idx >= rows * cols) return;
    const int r = idx / cols;
    const int c = idx - r * cols;
    dst[idx] = f2bfu(src[(size_t)r * sld + soff + c]);
}

__global__ __launch_bounds__(256) void zero_u16_kernel(unsigned short* p, int n) {
    const int idx = (int)(blockIdx.x * 256u + threadIdx.x);
    if (idx < n) p[idx] = 0;
}

// ---------------------------------------------------------------------------
// Host driver
// ---------------------------------------------------------------------------
extern "C" void kernel_launch(void* const* d_in, const int* in_sizes, int n_in,
                              void* d_out, int out_size, void* d_ws, size_t ws_size,
                              hipStream_t stream) {
    (void)in_sizes; (void)n_in; (void)out_size; (void)ws_size;
    const int*   node_types = (const int*)  d_in[0];
    const float* emb        = (const float*)d_in[1];
    const float* Wc_w       = (const float*)d_in[2];   // 512 x 576
    const float* Wc_b       = (const float*)d_in[3];
    const float* W_ih       = (const float*)d_in[4];   // 1536 x 512
    const float* W_hh       = (const float*)d_in[5];
    const float* b_ih       = (const float*)d_in[6];
    const float* b_hh       = (const float*)d_in[7];
    float* out = (float*)d_out;

    // ---- workspace carve-up (256B aligned) ----
    char*  ws  = (char*)d_ws;
    size_t off = 0;
    auto carve = [&](size_t bytes) -> void* {
        void* p = ws + off;
        off += (bytes + 255) & ~(size_t)255;
        return p;
    };
    unsigned short* wih_b = (unsigned short*)carve((size_t)G_DIM * H_DIM * 2);
    unsigned short* whh_b = (unsigned short*)carve((size_t)G_DIM * H_DIM * 2);
    unsigned short* wce_b = (unsigned short*)carve((size_t)H_DIM * E_DIM * 2);
    unsigned short* wch_b = (unsigned short*)carve((size_t)H_DIM * H_DIM * 2);
    unsigned short* emb_b = (unsigned short*)carve((size_t)20 * E_DIM * 2);
    unsigned short* bufH  = (unsigned short*)carve((size_t)32768 * H_DIM * 2);
    unsigned short* bufS  = (unsigned short*)carve((size_t)4096 * H_DIM * 2);
    float*          bufGX = (float*)carve((size_t)4096 * G_DIM * 4);
    float*          bufGH = (float*)carve((size_t)4096 * G_DIM * 4);

    auto cdiv = [](int a, int b) { return (a + b - 1) / b; };

    // ---- weight / emb conversion to bf16 ----
    {
        int n = G_DIM * H_DIM;
        convert_bf16_kernel<<<cdiv(n, 256), 256, 0, stream>>>(W_ih, H_DIM, 0, wih_b, G_DIM, H_DIM);
        convert_bf16_kernel<<<cdiv(n, 256), 256, 0, stream>>>(W_hh, H_DIM, 0, whh_b, G_DIM, H_DIM);
        n = H_DIM * E_DIM;
        convert_bf16_kernel<<<cdiv(n, 256), 256, 0, stream>>>(Wc_w, E_DIM + H_DIM, 0, wce_b, H_DIM, E_DIM);
        n = H_DIM * H_DIM;
        convert_bf16_kernel<<<cdiv(n, 256), 256, 0, stream>>>(Wc_w, E_DIM + H_DIM, E_DIM, wch_b, H_DIM, H_DIM);
        n = 20 * E_DIM;
        convert_bf16_kernel<<<cdiv(n, 256), 256, 0, stream>>>(emb, E_DIM, 0, emb_b, 20, E_DIM);
    }

    auto gemm = [&](const unsigned short* A1, int mul, int add,
                    const int* gather, const unsigned short* gbase,
                    int K1, const unsigned short* W1,
                    const unsigned short* A2, int K2, const unsigned short* W2,
                    const float* bias, float* oF, unsigned short* oB,
                    int M, int N, int act) {
        const int tiles  = ((M + 31) / 32) * (N / 64);
        const int blocks = (tiles + 7) / 8;   // 8 waves / 256-thread block
        wmma_gemm_kernel<<<blocks, 256, 0, stream>>>(
            A1, mul, add, gather, gbase, K1, W1, A2, K2, W2,
            bias, oF, oB, M, N, act);
    };

    // ---- leaves: h = tanh(emb[nt] @ Wc_e^T + Wc_b) ----
    gemm(nullptr, 0, 0, node_types + OFFSETS[DEPTH], emb_b, E_DIM, wce_b,
         nullptr, 0, nullptr, Wc_b, nullptr, bufH, SIZES[DEPTH], H_DIM, 1);

    // ---- levels d = 4 .. 0 ----
    for (int d = DEPTH - 1; d >= 0; --d) {
        const int n = SIZES[d];

        // h0 = 0
        zero_u16_kernel<<<cdiv(n * H_DIM, 256), 256, 0, stream>>>(bufS, n * H_DIM);

        for (int c = 0; c < C_CH; ++c) {
            // gx_c = seq[:, c, :] @ W_ih^T + b_ih  (row = m*8 + c of bufH)
            gemm(bufH, C_CH, c, nullptr, nullptr, H_DIM, wih_b,
                 nullptr, 0, nullptr, b_ih, bufGX, nullptr, n, G_DIM, 0);
            // gh = h @ W_hh^T + b_hh
            gemm(bufS, 1, 0, nullptr, nullptr, H_DIM, whh_b,
                 nullptr, 0, nullptr, b_hh, bufGH, nullptr, n, G_DIM, 0);
            // gate mixing (in-place on bufS)
            gru_gates_kernel<<<cdiv(n * H_DIM, 256), 256, 0, stream>>>(
                bufGX, bufGH, bufS, n * H_DIM);
        }

        // combine: h = tanh(emb[nt] @ Wc_e^T + h_state @ Wc_h^T + Wc_b)
        if (d > 0) {
            gemm(nullptr, 0, 0, node_types + OFFSETS[d], emb_b, E_DIM, wce_b,
                 bufS, H_DIM, wch_b, Wc_b, nullptr, bufH, n, H_DIM, 1);
        } else {
            gemm(nullptr, 0, 0, node_types + OFFSETS[0], emb_b, E_DIM, wce_b,
                 bufS, H_DIM, wch_b, Wc_b, out, nullptr, 1, H_DIM, 1);
        }
    }
}
